// GRUStrategyNetwork_73607149519333
// MI455X (gfx1250) — matrix operand order repaired and checked by reading, not compiled, not run on hardware
//
#include <hip/hip_runtime.h>
#include <hip/hip_bf16.h>

// ---------------------------------------------------------------------------
// Problem constants (from reference)
// ---------------------------------------------------------------------------
#define BB   2048
#define TT   64
#define DD   512
#define SS   256
#define PP   11
#define H1   384
#define DS_  768          // D + S
#define KX   544          // W_ih K padded 513 -> 544 (multiple of 32)
#define BM   16           // batch rows per workgroup

// LDS strides (in halfs) padded to dodge bank conflicts, 16B-aligned rows
#define CSTR 776          // combined [16][768]  (feat | h)
#define HSTR 392          // hidden   [16][384]
#define XSTR 40           // xtail    [16][32]   (pos | zeros)

typedef __attribute__((ext_vector_type(16))) __bf16    v16bf;
typedef __attribute__((ext_vector_type(8)))  float     v8f;
typedef __attribute__((ext_vector_type(4)))  unsigned  u32x4;

union ABFrag { u32x4 q[2]; v16bf v; };

__device__ __forceinline__ unsigned short f2bf(float x) {
  unsigned u = __float_as_uint(x);
  unsigned r = u + 0x7FFFu + ((u >> 16) & 1u);   // round-to-nearest-even
  return (unsigned short)(r >> 16);
}
__device__ __forceinline__ float bf2f(unsigned short h) {
  return __uint_as_float(((unsigned)h) << 16);
}

// A fragment: 16x32 bf16 tile, rows in lanes. ISA layout:
// lanes 0-15 hold K= kk+0..7 (v0-3) and kk+16..23 (v4-7); lanes 16-31 the +8 halves.
__device__ __forceinline__ v16bf loadA(const unsigned short* base, int stride,
                                       int kk, int lane) {
  int m  = lane & 15;
  int ko = (lane >> 4) * 8;
  const unsigned short* p = base + m * stride + kk + ko;
  ABFrag f;
  f.q[0] = *(const u32x4*)(p);
  f.q[1] = *(const u32x4*)(p + 16);
  return f.v;
}

// B fragment: 32x16 bf16 tile from row-major weight [N][K]:
// lane n holds column n0+(lane&15), K = kk + (lane>>4)*8 + 0..7 and +16..23 (matches A).
__device__ __forceinline__ v16bf loadB(const unsigned short* W, int ldb,
                                       int n0, int kk, int lane) {
  int n  = n0 + (lane & 15);
  int kb = kk + (lane >> 4) * 8;
  const unsigned short* p = W + (size_t)n * ldb + kb;
  ABFrag f;
  f.q[0] = *(const u32x4*)(p);
  f.q[1] = *(const u32x4*)(p + 16);
  return f.v;
}

__device__ __forceinline__ v8f bfwmma(v16bf a, v16bf b, v8f c) {
  return __builtin_amdgcn_wmma_f32_16x16x32_bf16(false, a, false, b,
                                                 (short)0, c, false, false);
}

// ---------------------------------------------------------------------------
// Weight prep: f32 -> bf16 (+ K-pad W_ih 513->544, N-pad W2 11->16)
// ---------------------------------------------------------------------------
__global__ void prep_weights(const float* __restrict__ W1,
                             const float* __restrict__ W_ih,
                             const float* __restrict__ W_hh,
                             const float* __restrict__ W2,
                             unsigned short* __restrict__ w1b,
                             unsigned short* __restrict__ wihb,
                             unsigned short* __restrict__ whhb,
                             unsigned short* __restrict__ w2b) {
  int i = blockIdx.x * blockDim.x + threadIdx.x;
  const int N1 = H1 * DS_;        // 294912
  const int N2 = 3 * SS * KX;     // 417792
  const int N3 = 3 * SS * SS;     // 196608
  const int N4 = 16 * H1;         // 6144 (W2 padded to 16 rows)
  if (i < N1) {
    w1b[i] = f2bf(W1[i]);
  } else if (i < N1 + N2) {
    int j = i - N1;
    int r = j / KX, c = j - r * KX;
    wihb[j] = (c < DD + 1) ? f2bf(W_ih[r * (DD + 1) + c]) : (unsigned short)0;
  } else if (i < N1 + N2 + N3) {
    int j = i - N1 - N2;
    whhb[j] = f2bf(W_hh[j]);
  } else if (i < N1 + N2 + N3 + N4) {
    int j = i - N1 - N2 - N3;
    int r = j / H1, c = j - r * H1;
    w2b[j] = (r < PP) ? f2bf(W2[r * H1 + c]) : (unsigned short)0;
  }
}

// ---------------------------------------------------------------------------
// Main persistent GRU kernel: 1 workgroup = 16 batch rows, loops all T steps
// ---------------------------------------------------------------------------
__global__ __launch_bounds__(256)
void gru_main(const float* __restrict__ price,          // [B,T,D]
              const float* __restrict__ uu,             // [B,T,P]
              const unsigned short* __restrict__ w1b,   // [H1][768] bf16
              const float* __restrict__ b1,             // [H1]
              const unsigned short* __restrict__ w2b,   // [16][384] bf16 (padded)
              const float* __restrict__ b2,             // [P]
              const float* __restrict__ temperature,    // [1]
              const unsigned short* __restrict__ wihb,  // [768][544] bf16
              const float* __restrict__ b_ih,           // [768]
              const unsigned short* __restrict__ whhb,  // [768][256] bf16
              const float* __restrict__ b_hh,           // [768]
              const float* __restrict__ h0,             // [S]
              float* __restrict__ out_pos,              // [B,T]
              float* __restrict__ out_states,           // [B,T,S]
              float* __restrict__ out_hf)               // [B,S]
{
  __shared__ unsigned short combined[BM * CSTR];  // bf16: feat(0..511) | h(512..767)
  __shared__ unsigned short hiddenS [BM * HSTR];  // bf16
  __shared__ unsigned short xtail   [BM * XSTR];  // bf16: pos | zeros
  __shared__ float          logitsS [BM * 12];
  __shared__ int            zeroS;                // opaque 0 (anti-LICM)

  const int tid  = threadIdx.x;
  const int lane = tid & 31;
  const int w    = tid >> 5;          // wave 0..7
  const int m0   = blockIdx.x * BM;   // batch tile base
  const float temp = temperature[0];

  // ---- init: h := h0 broadcast (bf16), zero xtail, zero opaque scalar ----
  {
    int m  = tid >> 4;
    int s0 = (tid & 15) * 16;
    #pragma unroll
    for (int i = 0; i < 16; ++i)
      combined[m * CSTR + DD + s0 + i] = f2bf(h0[s0 + i]);
  }
  if (tid < 160) {
    int base = tid * 4;
    #pragma unroll
    for (int i = 0; i < 4; ++i) xtail[base + i] = 0;
  }
  if (tid == 0) zeroS = 0;
  __syncthreads();

  for (int t = 0; t < TT; ++t) {
    // Opaque zero re-read each timestep: the compiler cannot prove the weight
    // addresses loop-invariant, so B-fragment loads stay inside the t-loop
    // (fresh global_load_b128 from L2-hot weights) instead of being hoisted
    // and spilled to scratch.
    const int tz = *(volatile int*)&zeroS;
    const unsigned short* w1t  = w1b  + tz;
    const unsigned short* w2t  = w2b  + tz;
    const unsigned short* wiht = wihb + tz;
    const unsigned short* whht = whhb + tz;

    // ---- 1. stage feature tile 16x512 f32 -> bf16 LDS (+ prefetch t+1) ----
    {
      int m  = tid >> 4;
      int c0 = (tid & 15) * 32;
      const float* src = price + ((size_t)(m0 + m) * TT + t) * DD + c0;
      unsigned short* dst = combined + m * CSTR + c0;
      #pragma unroll
      for (int i = 0; i < 8; ++i) {
        float4 v = ((const float4*)src)[i];
        dst[i * 4 + 0] = f2bf(v.x);
        dst[i * 4 + 1] = f2bf(v.y);
        dst[i * 4 + 2] = f2bf(v.z);
        dst[i * 4 + 3] = f2bf(v.w);
      }
      if (t + 1 < TT)
        __builtin_prefetch(price + ((size_t)(m0 + m) * TT + (t + 1)) * DD + c0, 0, 1);
    }
    __syncthreads();

    // ---- 2. GEMM1: hidden = relu(combined @ W1^T + b1); 3 N-tiles / wave ----
    #pragma unroll
    for (int jt = 0; jt < 3; ++jt) {
      int n0 = (w * 3 + jt) * 16;
      v8f acc = {};
      for (int kk = 0; kk < DS_; kk += 32) {
        v16bf a = loadA(combined, CSTR, kk, lane);
        v16bf b = loadB(w1t, DS_, n0, kk, lane);
        acc = bfwmma(a, b, acc);
      }
      int n  = n0 + (lane & 15);
      int mb = (lane >> 4) * 8;
      float bias = b1[n];
      #pragma unroll
      for (int r = 0; r < 8; ++r) {
        float v = acc[r] + bias;
        v = v > 0.0f ? v : 0.0f;
        hiddenS[(mb + r) * HSTR + n] = f2bf(v);
      }
    }
    __syncthreads();

    // ---- 3. logits = hidden @ W2^T + b2: one WMMA tile on wave 0
    //         (W2 N-padded 11->16 with zero rows) ----
    if (w == 0) {
      v8f acc = {};
      for (int kk = 0; kk < H1; kk += 32) {
        v16bf a = loadA(hiddenS, HSTR, kk, lane);
        v16bf b = loadB(w2t, H1, 0, kk, lane);
        acc = bfwmma(a, b, acc);
      }
      int p  = lane & 15;
      int mb = (lane >> 4) * 8;
      float bias = b2[p < PP ? p : 0];
      if (p < PP) {
        #pragma unroll
        for (int r = 0; r < 8; ++r)
          logitsS[(mb + r) * 12 + p] = acc[r] + bias;
      }
    }
    __syncthreads();

    // ---- 4. Gumbel-softmax -> expected position (16 threads) ----
    if (tid < BM) {
      int m = tid;
      const float* up = uu + ((size_t)(m0 + m) * TT + t) * PP;
      float l[PP];
      float mx = -1e30f;
      for (int p = 0; p < PP; ++p) {
        float g = -logf(-logf(up[p] + 1e-20f) + 1e-20f);
        l[p] = (logitsS[m * 12 + p] + g) / temp;
        mx = fmaxf(mx, l[p]);
      }
      float se = 0.0f, sw = 0.0f;
      for (int p = 0; p < PP; ++p) {
        float e = expf(l[p] - mx);
        se += e; sw += e * (float)p;
      }
      float pos = sw / se;
      out_pos[(size_t)(m0 + m) * TT + t] = pos;
      xtail[m * XSTR] = f2bf(pos);
    }
    __syncthreads();

    // ---- 5. GEMM3/4: gi = x@W_ih^T, gh = h@W_hh^T; 2 state-tiles/wave,
    //         3 gates each, all accumulators in registers -----------------
    v8f aI[2][3], aH[2][3];
    #pragma unroll
    for (int jt = 0; jt < 2; ++jt) {
      int st = (w * 2 + jt) * 16;          // state-column tile base
      #pragma unroll
      for (int g = 0; g < 3; ++g) { aI[jt][g] = {}; aH[jt][g] = {}; }
      // gi over feat part (K = 0..511)
      for (int kk = 0; kk < DD; kk += 32) {
        v16bf a = loadA(combined, CSTR, kk, lane);
        #pragma unroll
        for (int g = 0; g < 3; ++g) {
          v16bf b = loadB(wiht, KX, g * SS + st, kk, lane);
          aI[jt][g] = bfwmma(a, b, aI[jt][g]);
        }
      }
      // gi tail (K = 512..543: pos | zeros)
      {
        v16bf a = loadA(xtail, XSTR, 0, lane);
        #pragma unroll
        for (int g = 0; g < 3; ++g) {
          v16bf b = loadB(wiht, KX, g * SS + st, DD, lane);
          aI[jt][g] = bfwmma(a, b, aI[jt][g]);
        }
      }
      // gh over h (K = 0..255, read from combined's h region)
      for (int kk = 0; kk < SS; kk += 32) {
        v16bf a = loadA(combined, CSTR, DD + kk, lane);
        #pragma unroll
        for (int g = 0; g < 3; ++g) {
          v16bf b = loadB(whht, SS, g * SS + st, kk, lane);
          aH[jt][g] = bfwmma(a, b, aH[jt][g]);
        }
      }
    }
    __syncthreads();   // every wave finished reading h before anyone rewrites it

    // ---- 6. GRU gates + state update (in-register) ----
    #pragma unroll
    for (int jt = 0; jt < 2; ++jt) {
      int st = (w * 2 + jt) * 16;
      int s  = st + (lane & 15);
      int mb = (lane >> 4) * 8;
      float bIr = b_ih[s], bIz = b_ih[SS + s], bIn = b_ih[2 * SS + s];
      float bHr = b_hh[s], bHz = b_hh[SS + s], bHn = b_hh[2 * SS + s];
      #pragma unroll
      for (int r = 0; r < 8; ++r) {
        int m = mb + r;
        float hprev = bf2f(combined[m * CSTR + DD + s]);
        float rg = 1.0f / (1.0f + expf(-(aI[jt][0][r] + bIr + aH[jt][0][r] + bHr)));
        float zg = 1.0f / (1.0f + expf(-(aI[jt][1][r] + bIz + aH[jt][1][r] + bHz)));
        float ng = tanhf(aI[jt][2][r] + bIn + rg * (aH[jt][2][r] + bHn));
        float hn = (1.0f - zg) * ng + zg * hprev;
        combined[m * CSTR + DD + s] = f2bf(hn);
        out_states[((size_t)(m0 + m) * TT + t) * SS + s] = hn;
        if (t == TT - 1) out_hf[(size_t)(m0 + m) * SS + s] = hn;
      }
    }
    __syncthreads();
  }
}

// ---------------------------------------------------------------------------
// Host launcher
// ---------------------------------------------------------------------------
extern "C" void kernel_launch(void* const* d_in, const int* in_sizes, int n_in,
                              void* d_out, int out_size, void* d_ws, size_t ws_size,
                              hipStream_t stream) {
  const float* price = (const float*)d_in[0];
  const float* uu    = (const float*)d_in[1];
  const float* W1    = (const float*)d_in[2];
  const float* b1    = (const float*)d_in[3];
  const float* W2    = (const float*)d_in[4];
  const float* b2    = (const float*)d_in[5];
  const float* temp  = (const float*)d_in[6];
  const float* W_ih  = (const float*)d_in[7];
  const float* b_ih  = (const float*)d_in[8];
  const float* W_hh  = (const float*)d_in[9];
  const float* b_hh  = (const float*)d_in[10];
  const float* h0    = (const float*)d_in[11];

  // workspace layout (bf16 weights), all 16B-aligned
  unsigned short* w1b  = (unsigned short*)d_ws;                       // 384*768
  unsigned short* wihb = w1b + (size_t)H1 * DS_;                      // 768*544
  unsigned short* whhb = wihb + (size_t)3 * SS * KX;                  // 768*256
  unsigned short* w2b  = whhb + (size_t)3 * SS * SS;                  // 16*384

  float* out_pos    = (float*)d_out;                                  // [B,T]
  float* out_states = out_pos + (size_t)BB * TT;                      // [B,T,S]
  float* out_hf     = out_states + (size_t)BB * TT * SS;              // [B,S]

  const int prep_total = H1 * DS_ + 3 * SS * KX + 3 * SS * SS + 16 * H1;
  prep_weights<<<(prep_total + 255) / 256, 256, 0, stream>>>(
      W1, W_ih, W_hh, W2, w1b, wihb, whhb, w2b);

  gru_main<<<BB / BM, 256, 0, stream>>>(
      price, uu, w1b, b1, w2b, b2, temp, wihb, b_ih, whhb, b_hh, h0,
      out_pos, out_states, out_hf);
}
